// PointNetFeat_22677427323460
// MI455X (gfx1250) — matrix operand order repaired and checked by reading, not compiled, not run on hardware
//
#include <hip/hip_runtime.h>

// ---------------------------------------------------------------------------
// Frustum-ConvNet PointNet stage for MI455X (gfx1250, wave32, WMMA).
//
// Per module i (M,K,dist):
//   1) query_gather: wave-per-query ballot scan of pc z, first-K ordered hits,
//      builds X0 [P, 32] f16 (channels 0..3 = local xyz + feat, 4..31 = 0)
//   2) 3x { zero stats; WMMA GEMM Y=W*X (f16 in, f32 acc, f16 out) + per-
//      channel sum/sumsq atomics; BN(train-stats)+ReLU normalize IN PLACE }
//      (conv bias is dropped: it cancels exactly under batch-stat BN)
//   3) masked max-pool over K -> d_out (float, [B, C, M])
//
// GEMM v2: block = 8 waves = 64co x 256p macro tile. W slab (64 x Cin f16,
// <= 32 KB) staged in LDS (async global->LDS when available), A fragments
// read via ds_load_b128; each wave register-blocks 32co x 64p = 8 WMMA
// accumulators, 8 v_wmma per 32-deep k-step.
// ---------------------------------------------------------------------------

typedef _Float16 half8 __attribute__((ext_vector_type(8)));
typedef _Float16 v16h  __attribute__((ext_vector_type(16)));
typedef float    v8f   __attribute__((ext_vector_type(8)));

#define NPTS 16384
#define BATCH 8
#define BN_EPS 1e-5f

#if defined(__has_builtin)
#if __has_builtin(__builtin_amdgcn_global_load_async_to_lds_b128) && \
    __has_builtin(__builtin_amdgcn_s_wait_asynccnt)
#define FCN_ASYNC_LDS 1
#endif
#endif

// ---------------------------------------------------------------- zero stats
__global__ void fcn_zero_f32(float* __restrict__ p, int n) {
  int i = blockIdx.x * blockDim.x + threadIdx.x;
  if (i < n) p[i] = 0.0f;
}

// ------------------------------------------------------- weight f32 -> f16
// Wsrc: [Cout, Cin] f32 row-major  ->  Wdst: [Cout, Cinpad] f16 (zero pad)
__global__ void fcn_convert_w(const float* __restrict__ Wsrc,
                              _Float16* __restrict__ Wdst,
                              int Cin, int Cinpad, int total) {
  int i = blockIdx.x * blockDim.x + threadIdx.x;
  if (i >= total) return;
  int c = i % Cinpad;
  int r = i / Cinpad;
  float v = (c < Cin) ? Wsrc[r * Cin + c] : 0.0f;
  Wdst[i] = (_Float16)v;
}

// --------------------------------------------------- query + gather (wave32)
// One wave per query (b,m). Ballot-scan z for |z - qz| < dist, first-K hits in
// index order fill X0 rows; empty slots pre-filled with point 0 (= reference's
// idx padding with 0). flags[q] = (any hit).
__global__ void __launch_bounds__(128)
fcn_query_gather(const float* __restrict__ pc,     // [B,3,N]
                 const float* __restrict__ feat,   // [B,1,N]
                 const float* __restrict__ ctr,    // [B,3,M]
                 _Float16* __restrict__ X0,        // [B*M*K, 32]
                 float* __restrict__ flags,        // [B*M]
                 int M, int K, float dist) {
  int q = blockIdx.x * 4 + (threadIdx.x >> 5);
  if (q >= BATCH * M) return;
  int lane = threadIdx.x & 31;
  int b = q / M, m = q - b * M;

  const float* pcb = pc + (size_t)b * 3 * NPTS;
  const float* fb  = feat + (size_t)b * NPTS;
  float qx = ctr[((size_t)b * 3 + 0) * M + m];
  float qy = ctr[((size_t)b * 3 + 1) * M + m];
  float qz = ctr[((size_t)b * 3 + 2) * M + m];

  size_t base = (size_t)q * K;

  // default fill with point 0 (reference pads idx with 0)
  float dx0 = pcb[0] - qx, dy0 = pcb[NPTS] - qy, dz0 = pcb[2 * NPTS] - qz;
  float f0 = fb[0];
  for (int s = lane; s < K; s += 32) {
    _Float16* xp = X0 + (base + s) * 32;
    xp[0] = (_Float16)dx0; xp[1] = (_Float16)dy0;
    xp[2] = (_Float16)dz0; xp[3] = (_Float16)f0;
#pragma unroll
    for (int c = 4; c < 32; ++c) xp[c] = (_Float16)0.0f;
  }

  int cnt = 0;
  for (int n0 = 0; n0 < NPTS; n0 += 32) {
    int n = n0 + lane;
    float z = pcb[2 * NPTS + n];
    bool hit = __builtin_fabsf(z - qz) < dist;
    unsigned msk = __builtin_amdgcn_ballot_w32(hit);   // wave32 ballot
    if (hit) {
      int slot = cnt + __popc(msk & ((1u << lane) - 1u));
      if (slot < K) {
        _Float16* xp = X0 + (base + slot) * 32;
        xp[0] = (_Float16)(pcb[n] - qx);
        xp[1] = (_Float16)(pcb[NPTS + n] - qy);
        xp[2] = (_Float16)(z - qz);
        xp[3] = (_Float16)fb[n];
      }
    }
    cnt += __popc(msk);          // wave-uniform
    if (cnt >= K) break;         // uniform early exit
  }
  if (lane == 0) flags[q] = (cnt > 0) ? 1.0f : 0.0f;
}

// -------------------------------------------------------- WMMA GEMM + stats
// Y[p, co] = sum_c W[co, c] * X[p, c]   (f16 A/B, f32 accumulate, f16 out)
// W: [Cout, Cin] f16 row-major, X: [P, Cin] f16 row-major, Y: [P, Cout] f16.
// Block tile: 64 Cout x 256 P. Wave tile: 32 Cout x 64 P (8 accumulators).
// W slab for the block's 64 rows lives in LDS.
__global__ void __launch_bounds__(256)
fcn_gemm_wmma(const _Float16* __restrict__ W,
              const _Float16* __restrict__ X,
              _Float16* __restrict__ Y,
              float* __restrict__ stats,   // [Cout] sum, [Cout] sumsq
              int Cin, int Cout, int P) {
  __shared__ _Float16 sW[64 * 256];   // 32 KB max (Cin <= 256)
  __shared__ float s_sum[64];
  __shared__ float s_sq[64];
  if (threadIdx.x < 64) { s_sum[threadIdx.x] = 0.0f; s_sq[threadIdx.x] = 0.0f; }

  // ---- stage W[coBase .. coBase+63][0..Cin) into LDS (contiguous slab) ----
  int coBase = blockIdx.y * 64;
  const _Float16* wsrc = W + (size_t)coBase * Cin;
  int n128 = (64 * Cin) >> 3;          // number of half8 chunks
#if defined(FCN_ASYNC_LDS)
  {
    typedef int v4i_t __attribute__((vector_size(16)));
    typedef __attribute__((address_space(1))) v4i_t* gvp_t;
    typedef __attribute__((address_space(3))) v4i_t* lvp_t;
    for (int idx = threadIdx.x; idx < n128; idx += 256) {
      __builtin_amdgcn_global_load_async_to_lds_b128(
          (gvp_t)(void*)(wsrc + (size_t)idx * 8),
          (lvp_t)(sW + (size_t)idx * 8), 0, 0);
    }
    __builtin_amdgcn_s_wait_asynccnt(0);
  }
#else
  for (int idx = threadIdx.x; idx < n128; idx += 256) {
    ((half8*)sW)[idx] = ((const half8*)wsrc)[idx];
  }
#endif
  __syncthreads();

  int wave = threadIdx.x >> 5;
  int lane = threadIdx.x & 31;
  int r  = lane & 15;       // column-within-tile (B/C/D N index), A row
  int hi = lane >> 4;       // half-wave select
  int cop = wave >> 2;      // 0..1 : which 32-row co group
  int pw  = wave & 3;       // 0..3 : which 64-position strip
  int coW = cop * 32;       // local co offset of this wave's group
  int p0  = blockIdx.x * 256 + pw * 64;

  v8f acc[2][4];
#pragma unroll
  for (int ci = 0; ci < 2; ++ci)
#pragma unroll
    for (int t = 0; t < 4; ++t) acc[ci][t] = (v8f){};

  for (int kc = 0; kc < Cin; kc += 32) {
    // A fragments from LDS: rows coW + ci*16 + r
    v16h a[2];
#pragma unroll
    for (int ci = 0; ci < 2; ++ci) {
      const _Float16* ar = sW + (size_t)(coW + ci * 16 + r) * Cin + kc + hi * 8;
      half8 alo = *(const half8*)(ar);
      half8 ahi = *(const half8*)(ar + 16);
      a[ci] = __builtin_shufflevector(alo, ahi, 0, 1, 2, 3, 4, 5, 6, 7,
                                      8, 9, 10, 11, 12, 13, 14, 15);
    }
    // B fragments from global: col N = p0 + 16t + r
#pragma unroll
    for (int t = 0; t < 4; ++t) {
      const _Float16* xr = X + (size_t)(p0 + 16 * t + r) * Cin + kc + hi * 8;
      half8 blo = *(const half8*)(xr);
      half8 bhi = *(const half8*)(xr + 16);
      v16h bf = __builtin_shufflevector(blo, bhi, 0, 1, 2, 3, 4, 5, 6, 7,
                                        8, 9, 10, 11, 12, 13, 14, 15);
      acc[0][t] = __builtin_amdgcn_wmma_f32_16x16x32_f16(
          false, a[0], false, bf, (short)0, acc[0][t], false, false);
      acc[1][t] = __builtin_amdgcn_wmma_f32_16x16x32_f16(
          false, a[1], false, bf, (short)0, acc[1][t], false, false);
    }
  }

  // D layout: lane holds column p0+16t+r, channels coBase+coW+ci*16+hi*8+{0..7}
#pragma unroll
  for (int ci = 0; ci < 2; ++ci) {
#pragma unroll
    for (int t = 0; t < 4; ++t) {
      half8 h;
#pragma unroll
      for (int i = 0; i < 8; ++i) h[i] = (_Float16)acc[ci][t][i];
      *(half8*)(Y + (size_t)(p0 + 16 * t + r) * Cout +
                coBase + coW + ci * 16 + hi * 8) = h;
    }
  }

  // per-channel partial sums over this wave's 64 columns
#pragma unroll
  for (int ci = 0; ci < 2; ++ci) {
#pragma unroll
    for (int i = 0; i < 8; ++i) {
      float v = 0.0f, s = 0.0f;
#pragma unroll
      for (int t = 0; t < 4; ++t) {
        float y = acc[ci][t][i];
        v += y;
        s += y * y;
      }
#pragma unroll
      for (int off = 1; off < 16; off <<= 1) {
        v += __shfl_xor(v, off, 32);
        s += __shfl_xor(s, off, 32);
      }
      if (r == 0) {
        atomicAdd(&s_sum[coW + ci * 16 + hi * 8 + i], v);
        atomicAdd(&s_sq[coW + ci * 16 + hi * 8 + i], s);
      }
    }
  }
  __syncthreads();
  if (threadIdx.x < 64) {
    atomicAdd(&stats[coBase + threadIdx.x], s_sum[threadIdx.x]);
    atomicAdd(&stats[Cout + coBase + threadIdx.x], s_sq[threadIdx.x]);
  }
}

// ---------------------------------------------- BN (batch stats) + ReLU
// In place on Y [P, Cout] f16. mean/var from stats; gamma/beta f32.
__global__ void fcn_bn_relu(_Float16* __restrict__ Y,
                            const float* __restrict__ stats,
                            const float* __restrict__ gamma,
                            const float* __restrict__ beta,
                            int Cout, int total, float invP) {
  int i = blockIdx.x * blockDim.x + threadIdx.x;
  if (i >= total) return;
  int c = i % Cout;
  float mean = stats[c] * invP;
  float var  = stats[Cout + c] * invP - mean * mean;
  float rs   = rsqrtf(var + BN_EPS);
  float y = (float)Y[i];
  y = (y - mean) * rs * gamma[c] + beta[c];
  Y[i] = (_Float16)fmaxf(y, 0.0f);
}

// ----------------------------------------------------- masked max-pool over K
// X: [B*M*K, Cout] f16 (ReLU output >= 0) -> out[b, c, m] f32
__global__ void fcn_maxpool(const _Float16* __restrict__ X,
                            const float* __restrict__ flags,
                            float* __restrict__ out,
                            int Cout, int M, int K, int total) {
  int i = blockIdx.x * blockDim.x + threadIdx.x;
  if (i >= total) return;
  int c  = i % Cout;
  int bm = i / Cout;
  int b = bm / M, m = bm - b * M;
  const _Float16* xp = X + (size_t)bm * K * Cout + c;
  float mx = 0.0f;
  for (int k = 0; k < K; ++k) mx = fmaxf(mx, (float)xp[(size_t)k * Cout]);
  out[((size_t)b * Cout + c) * M + m] = mx * flags[bm];
}

// ---------------------------------------------------------------------------
extern "C" void kernel_launch(void* const* d_in, const int* in_sizes, int n_in,
                              void* d_out, int out_size, void* d_ws,
                              size_t ws_size, hipStream_t stream) {
  (void)in_sizes; (void)n_in; (void)out_size; (void)ws_size;

  static const int   Ms[4]    = {512, 256, 128, 64};
  static const int   Ksam[4]  = {32, 64, 64, 128};
  static const float dists[4] = {0.25f, 0.5f, 1.0f, 2.0f};
  static const int   mlp[4][4] = {{4, 64, 64, 128}, {4, 64, 64, 128},
                                  {4, 128, 128, 256}, {4, 256, 256, 512}};

  const float* pc   = (const float*)d_in[0];
  const float* feat = (const float*)d_in[1];
  const float* ctrs[4] = {(const float*)d_in[2], (const float*)d_in[3],
                          (const float*)d_in[4], (const float*)d_in[5]};
  // params pytree order per layer-dict (sorted keys): W, b, beta, gamma
  auto pW = [&](int i, int j) { return (const float*)d_in[6 + (i * 3 + j) * 4 + 0]; };
  auto pBeta  = [&](int i, int j) { return (const float*)d_in[6 + (i * 3 + j) * 4 + 2]; };
  auto pGamma = [&](int i, int j) { return (const float*)d_in[6 + (i * 3 + j) * 4 + 3]; };

  // workspace carve-up
  char* ws = (char*)d_ws;
  float*    stats  = (float*)ws;                         // 8 KB
  float*    flags  = (float*)(ws + 8 * 1024);            // 16 KB
  _Float16* warena = (_Float16*)(ws + 32 * 1024);        // < 1 MB of f16 W
  _Float16* X0 = (_Float16*)(ws + (size_t)(2)  * 1048576); //  8 MB  [P,32]
  _Float16* YA = (_Float16*)(ws + (size_t)(12) * 1048576); // 64 MB
  _Float16* YB = (_Float16*)(ws + (size_t)(76) * 1048576); // 32 MB

  // convert all 12 weight matrices to f16 (layer0 zero-padded Cin 4 -> 32)
  size_t woff[12];
  {
    size_t off = 0;
    for (int i = 0; i < 4; ++i)
      for (int j = 0; j < 3; ++j) {
        int cin = mlp[i][j], cinp = (j == 0) ? 32 : cin, cout = mlp[i][j + 1];
        woff[i * 3 + j] = off;
        int total = cout * cinp;
        fcn_convert_w<<<(total + 255) / 256, 256, 0, stream>>>(
            pW(i, j), warena + off, cin, cinp, total);
        off += (size_t)total;
      }
  }

  size_t out_off = 0;
  for (int i = 0; i < 4; ++i) {
    int M = Ms[i], K = Ksam[i];
    int P = BATCH * M * K;

    int nq = BATCH * M;
    fcn_query_gather<<<(nq + 3) / 4, 128, 0, stream>>>(
        pc, feat, ctrs[i], X0, flags, M, K, dists[i]);

    const _Float16* Xcur = X0;
    int Cin = 32;
    for (int j = 0; j < 3; ++j) {
      int Cout = mlp[i][j + 1];
      _Float16* Yb = (j & 1) ? YB : YA;

      fcn_zero_f32<<<(2 * Cout + 255) / 256, 256, 0, stream>>>(stats, 2 * Cout);

      dim3 grid(P / 256, Cout / 64, 1);
      fcn_gemm_wmma<<<grid, 256, 0, stream>>>(warena + woff[i * 3 + j], Xcur,
                                              Yb, stats, Cin, Cout, P);

      int total = P * Cout;
      fcn_bn_relu<<<(total + 255) / 256, 256, 0, stream>>>(
          Yb, stats, pGamma(i, j), pBeta(i, j), Cout, total, 1.0f / (float)P);

      Xcur = Yb;
      Cin = Cout;
    }

    int Cf = mlp[i][3];
    int total = BATCH * M * Cf;
    fcn_maxpool<<<(total + 255) / 256, 256, 0, stream>>>(
        Xcur, flags, (float*)d_out + out_off, Cf, M, K, total);
    out_off += (size_t)total;
  }
}
